// Torch_Ops_Aten_QuantizedGru_DataModule_53987738910818
// MI455X (gfx1250) — matrix operand order repaired and checked by reading, not compiled, not run on hardware
//
#include <hip/hip_runtime.h>
#include <cstdint>

// Problem constants (match reference)
#define TT 256
#define BB 64
#define II 2048
#define HH 1024
#define LL 2
#define GG (3 * HH)
#define TB (TT * BB)
#define OW (2 * HH)     // per-layer output width == II
#define LDA 2056        // padded LDS leading dim (2048 + 8 halves = +16B) for bank spread

typedef __attribute__((ext_vector_type(16))) _Float16 v16h;
typedef __attribute__((ext_vector_type(8)))  _Float16 v8h;
typedef __attribute__((ext_vector_type(8)))  float    v8f;
typedef __attribute__((ext_vector_type(4)))  int      v4i;

// ---- CDNA5 async global->LDS path (probe via __has_builtin; sync fallback) ----
#if defined(__has_builtin)
#if __has_builtin(__builtin_amdgcn_global_load_async_to_lds_b128) && \
    __has_builtin(__builtin_amdgcn_s_wait_asynccnt)
#define HAS_ASYNC_LDS 1
#endif
#endif

typedef __attribute__((address_space(1))) v4i* as1_v4i_p;
typedef __attribute__((address_space(3))) v4i* as3_v4i_p;

static __device__ __forceinline__ as1_v4i_p glb_v4i(const void* p) {
  return (as1_v4i_p)(uintptr_t)p;
}
static __device__ __forceinline__ as3_v4i_p lds_v4i(void* p) {
  // generic LDS pointer keeps the LDS offset in its low 32 bits
  return (as3_v4i_p)(uint32_t)(uintptr_t)p;
}

static __device__ __forceinline__ v16h cat8(v8h lo, v8h hi) {
  return __builtin_shufflevector(lo, hi, 0,1,2,3,4,5,6,7,8,9,10,11,12,13,14,15);
}

// A fragment (16xK=32, f16, row-major src, leading dim ld in halves).
// Lane l<16: M=l, K=[0..7],[16..23]; lane l>=16: M=l-16, K=[8..15],[24..31].
static __device__ __forceinline__ v16h load_a_frag(const _Float16* base, int ld, int lane) {
  int m  = lane & 15;
  int kh = (lane >> 4) << 3;                 // 0 or 8
  const _Float16* p = base + (size_t)m * ld + kh;
  v8h lo = *(const v8h*)p;
  v8h hi = *(const v8h*)(p + 16);
  return cat8(lo, hi);
}

// B fragment (K=32 x 16): B[k][n] = W[(n0+n)*ldw + k0 + k] (W row-major GxK).
// Lane n<16: col n, K=[0..7],[16..23]; lane n>=16: col n-16, K=[8..15],[24..31].
static __device__ __forceinline__ v16h load_b_frag(const _Float16* wbase, int ldw, int lane) {
  int n  = lane & 15;
  int kh = (lane >> 4) << 3;                 // 0 or 8
  const _Float16* p = wbase + (size_t)n * ldw + kh;
  v8h lo = *(const v8h*)p;
  v8h hi = *(const v8h*)(p + 16);
  return cat8(lo, hi);
}

static __device__ __forceinline__ float sigmoidf_(float x) {
  return 1.0f / (1.0f + __expf(-x));
}

// ---------------- weight dequant: int8 * scale -> f16 ----------------
__global__ __launch_bounds__(256) void dequant_wih_kernel(
    const int8_t* __restrict__ q, const float* __restrict__ scale,
    _Float16* __restrict__ out) {
  size_t i = (size_t)blockIdx.x * blockDim.x + threadIdx.x;   // < L*2*G*I
  int ld = (int)(i / ((size_t)GG * II));
  out[i] = (_Float16)((float)q[i] * scale[ld]);
}

__global__ __launch_bounds__(256) void dequant_whh_kernel(
    const int8_t* __restrict__ q, const float* __restrict__ scale,
    _Float16* __restrict__ out) {
  size_t i = (size_t)blockIdx.x * blockDim.x + threadIdx.x;   // < L*2*G*H
  int ld = (int)(i / ((size_t)GG * HH));
  out[i] = (_Float16)((float)q[i] * scale[ld]);
}

// ---------------- prefix sum of batch_sizes ----------------
__global__ void offsets_kernel(const int* __restrict__ bs, int* __restrict__ off) {
  if (threadIdx.x == 0 && blockIdx.x == 0) {
    int acc = 0;
    for (int t = 0; t < TT; ++t) { off[t] = acc; acc += bs[t]; }
    off[TT] = acc;
  }
}

// ---------------- unpack packed data -> dense (T,B,I) f16 ----------------
__global__ __launch_bounds__(256) void unpack_kernel(
    const float* __restrict__ data, const int* __restrict__ bs,
    const int* __restrict__ off, _Float16* __restrict__ x) {
  int tb = blockIdx.x;
  int t = tb / BB, b = tb % BB;
  bool m = b < bs[t];
  const float* src = data + (size_t)(off[t] + b) * II;
  _Float16* dst = x + (size_t)tb * II;
  for (int i = threadIdx.x; i < II; i += 256)
    dst[i] = m ? (_Float16)src[i] : (_Float16)0.0f;
}

// ---------------- gx = x @ W_ih^T + b_ih  (both directions) ----------------
// block = 128 threads (4 waves). The block's 16-row A panel (16x2048 f16,
// 64KB) is staged into LDS ONCE (async-to-LDS when available), then the
// barrier-free K loop streams B from global. Each wave computes 4 N-tiles,
// reusing the A fragment across 4 WMMAs per K step.
__global__ __launch_bounds__(128) void gemm_gx_kernel(
    const _Float16* __restrict__ x, const _Float16* __restrict__ wih,
    const float* __restrict__ bih, float* __restrict__ gx, int layer) {
  __shared__ __align__(16) _Float16 As[16 * LDA];
  const int d    = blockIdx.z;
  const int tb0  = blockIdx.x * 16;
  const int wave = threadIdx.x >> 5;
  const int lane = threadIdx.x & 31;
  const int g0   = blockIdx.y * 256 + wave * 64;   // 4 waves x 4 tiles = 256 cols
  const _Float16* W  = wih + (size_t)(layer * 2 + d) * GG * II;
  const float* bias  = bih + (size_t)(layer * 2 + d) * GG;

  // Stage A panel (contiguous 32768 halves in global; padded rows in LDS)
  {
    const _Float16* src = x + (size_t)tb0 * II;
    #pragma unroll
    for (int i = 0; i < 32; ++i) {
      int fh  = (i * 128 + threadIdx.x) * 8;       // 8 halves = 16B per op
      int row = fh >> 11, col = fh & 2047;
      _Float16* dst = As + row * LDA + col;
#if HAS_ASYNC_LDS
      __builtin_amdgcn_global_load_async_to_lds_b128(glb_v4i(src + fh), lds_v4i(dst), 0, 0);
#else
      *(v8h*)dst = *(const v8h*)(src + fh);
#endif
    }
#if HAS_ASYNC_LDS
    __builtin_amdgcn_s_wait_asynccnt(0);
#endif
    __syncthreads();
  }

  v8f acc0 = {}, acc1 = {}, acc2 = {}, acc3 = {};
  for (int k0 = 0; k0 < II; k0 += 32) {
    v16h a = load_a_frag(As + k0, LDA, lane);      // ds_load_b128 x2, no barriers
    const _Float16* wb = W + (size_t)g0 * II + k0;
    __builtin_prefetch(wb + 128, 0, 1);            // next K slab of B stream
    v16h b0 = load_b_frag(wb,                       II, lane);
    v16h b1 = load_b_frag(wb + (size_t)16 * II,     II, lane);
    v16h b2 = load_b_frag(wb + (size_t)32 * II,     II, lane);
    v16h b3 = load_b_frag(wb + (size_t)48 * II,     II, lane);
    acc0 = __builtin_amdgcn_wmma_f32_16x16x32_f16(false, a, false, b0, (short)0, acc0, false, false);
    acc1 = __builtin_amdgcn_wmma_f32_16x16x32_f16(false, a, false, b1, (short)0, acc1, false, false);
    acc2 = __builtin_amdgcn_wmma_f32_16x16x32_f16(false, a, false, b2, (short)0, acc2, false, false);
    acc3 = __builtin_amdgcn_wmma_f32_16x16x32_f16(false, a, false, b3, (short)0, acc3, false, false);
  }

  const int n = lane & 15;
  const int mbase = (lane >> 4) << 3;
  float* gxd = gx + (size_t)d * TB * GG;
  v8f accs[4] = {acc0, acc1, acc2, acc3};
  #pragma unroll
  for (int nt = 0; nt < 4; ++nt) {
    int g = g0 + nt * 16 + n;
    float bv = bias[g];
    #pragma unroll
    for (int v = 0; v < 8; ++v) {
      int m = v + mbase;
      gxd[(size_t)(tb0 + m) * GG + g] = accs[nt][v] + bv;
    }
  }
}

// ---------------- fused recurrent step (both directions) ----------------
// grid = (B/16, H/16, 2). One wave computes a 16x16 h-tile: three 16x16x1024
// gate GEMMs (96 WMMAs) + sigmoid/tanh gates + masked blend.
__global__ __launch_bounds__(32) void gru_step_kernel(
    const _Float16* __restrict__ hin16, const float* __restrict__ hin32,
    _Float16* __restrict__ hout16, float* __restrict__ hout32,
    const _Float16* __restrict__ whh,   // layer base: [2][G][H] f16
    const float* __restrict__ bhh,      // layer base: [2][G]
    const float* __restrict__ gx,       // [2][T*B][G]
    _Float16* __restrict__ xout,        // [T*B][2H] f16
    float* __restrict__ hn_out,         // d_out + total*2H : [4][B][H]
    const int* __restrict__ bs,
    int s, int last, int layer) {
  const int d  = blockIdx.z;
  const int tg = d ? (TT - 1 - s) : s;
  const int b0 = blockIdx.x * 16;
  const int j0 = blockIdx.y * 16;
  const int lane = threadIdx.x & 31;

  const _Float16* hA = hin16 + (size_t)d * BB * HH + (size_t)b0 * HH;
  const _Float16* Wd = whh + (size_t)d * GG * HH;

  v8f aR = {}, aZ = {}, aN = {};
  #pragma unroll 2
  for (int k0 = 0; k0 < HH; k0 += 32) {
    v16h a  = load_a_frag(hA + k0, HH, lane);
    v16h br = load_b_frag(Wd + (size_t)(j0)          * HH + k0, HH, lane);
    v16h bz = load_b_frag(Wd + (size_t)(HH + j0)     * HH + k0, HH, lane);
    v16h bn = load_b_frag(Wd + (size_t)(2 * HH + j0) * HH + k0, HH, lane);
    aR = __builtin_amdgcn_wmma_f32_16x16x32_f16(false, a, false, br, (short)0, aR, false, false);
    aZ = __builtin_amdgcn_wmma_f32_16x16x32_f16(false, a, false, bz, (short)0, aZ, false, false);
    aN = __builtin_amdgcn_wmma_f32_16x16x32_f16(false, a, false, bn, (short)0, aN, false, false);
  }

  const int j = j0 + (lane & 15);
  const float brb = bhh[(size_t)d * GG + j];
  const float bzb = bhh[(size_t)d * GG + HH + j];
  const float bnb = bhh[(size_t)d * GG + 2 * HH + j];
  const int bsz = bs[tg];
  const int mbase = (lane >> 4) << 3;
  const float* gxd = gx + (size_t)d * TB * GG;

  #pragma unroll
  for (int v = 0; v < 8; ++v) {
    int b = b0 + v + mbase;
    size_t tb = (size_t)tg * BB + b;
    const float* grow = gxd + tb * GG;
    float r  = sigmoidf_(grow[j]          + aR[v] + brb);
    float z  = sigmoidf_(grow[HH + j]     + aZ[v] + bzb);
    float nn = tanhf    (grow[2 * HH + j] + r * (aN[v] + bnb));
    float hp = hin32[(size_t)d * BB * HH + (size_t)b * HH + j];
    float hn = (1.0f - z) * nn + z * hp;
    if (b >= bsz) hn = hp;
    size_t hidx = (size_t)d * BB * HH + (size_t)b * HH + j;
    hout32[hidx] = hn;
    hout16[hidx] = (_Float16)hn;
    xout[tb * OW + (size_t)d * HH + j] = (_Float16)hn;
    if (last)
      hn_out[(size_t)(2 * layer + d) * BB * HH + (size_t)b * HH + j] = hn;
  }
}

// ---------------- init h buffers from hx ----------------
__global__ __launch_bounds__(256) void init_h_kernel(
    const float* __restrict__ hx, float* __restrict__ h32,
    _Float16* __restrict__ h16, int layer) {
  size_t i = (size_t)blockIdx.x * blockDim.x + threadIdx.x;   // < 2*B*H
  float v = hx[(size_t)(2 * layer) * BB * HH + i];
  h32[i] = v;
  h16[i] = (_Float16)v;
}

// ---------------- pack final layer output to f32 ----------------
__global__ __launch_bounds__(256) void pack_kernel(
    const _Float16* __restrict__ x, const int* __restrict__ bs,
    const int* __restrict__ off, float* __restrict__ out) {
  int tb = blockIdx.x;
  int t = tb / BB, b = tb % BB;
  if (b >= bs[t]) return;
  size_t row = (size_t)(off[t] + b);
  for (int c = threadIdx.x; c < OW; c += 256)
    out[row * OW + c] = (float)x[(size_t)tb * OW + c];
}

extern "C" void kernel_launch(void* const* d_in, const int* in_sizes, int n_in,
                              void* d_out, int out_size, void* d_ws, size_t ws_size,
                              hipStream_t stream) {
  const float*  data = (const float*)d_in[0];
  const int*    bs   = (const int*)d_in[1];
  const float*  hx   = (const float*)d_in[2];
  const int8_t* wihq = (const int8_t*)d_in[3];
  const int8_t* whhq = (const int8_t*)d_in[4];
  const float*  bih  = (const float*)d_in[5];
  const float*  bhh  = (const float*)d_in[6];
  const float*  sih  = (const float*)d_in[7];
  const float*  shh  = (const float*)d_in[8];
  float* out = (float*)d_out;
  const int total = in_sizes[0] / II;

  // Workspace carve (256B aligned slices)
  char* wp = (char*)d_ws;
  auto carve = [&](size_t bytes) {
    void* p = (void*)wp;
    wp += (bytes + 255) & ~(size_t)255;
    return p;
  };
  _Float16* xA    = (_Float16*)carve((size_t)TB * II * 2);
  _Float16* xB    = (_Float16*)carve((size_t)TB * OW * 2);
  _Float16* wih16 = (_Float16*)carve((size_t)LL * 2 * GG * II * 2);
  _Float16* whh16 = (_Float16*)carve((size_t)LL * 2 * GG * HH * 2);
  float*    gxbuf = (float*)   carve((size_t)2 * TB * GG * 4);
  float*    h32a  = (float*)   carve((size_t)2 * BB * HH * 4);
  float*    h32b  = (float*)   carve((size_t)2 * BB * HH * 4);
  _Float16* h16a  = (_Float16*)carve((size_t)2 * BB * HH * 2);
  _Float16* h16b  = (_Float16*)carve((size_t)2 * BB * HH * 2);
  int*      off   = (int*)     carve((size_t)(TT + 1) * 4);

  float*    h32buf[2] = {h32a, h32b};
  _Float16* h16buf[2] = {h16a, h16b};
  float*    hn_out = out + (size_t)total * OW;

  // 1) dequantize weights to f16
  {
    size_t n = (size_t)LL * 2 * GG * II;
    dequant_wih_kernel<<<(unsigned)(n / 256), 256, 0, stream>>>(wihq, sih, wih16);
    size_t m = (size_t)LL * 2 * GG * HH;
    dequant_whh_kernel<<<(unsigned)(m / 256), 256, 0, stream>>>(whhq, shh, whh16);
  }
  // 2) offsets + unpack
  offsets_kernel<<<1, 32, 0, stream>>>(bs, off);
  unpack_kernel<<<TB, 256, 0, stream>>>(data, bs, off, xA);

  // 3) layers
  _Float16* xin  = xA;
  _Float16* xout = xB;
  for (int l = 0; l < LL; ++l) {
    init_h_kernel<<<(2 * BB * HH) / 256, 256, 0, stream>>>(hx, h32buf[0], h16buf[0], l);
    gemm_gx_kernel<<<dim3(TB / 16, GG / 256, 2), 128, 0, stream>>>(
        xin, wih16, bih, gxbuf, l);
    for (int s = 0; s < TT; ++s) {
      int cur = s & 1, nxt = cur ^ 1;
      gru_step_kernel<<<dim3(BB / 16, HH / 16, 2), 32, 0, stream>>>(
          h16buf[cur], h32buf[cur], h16buf[nxt], h32buf[nxt],
          whh16 + (size_t)l * 2 * GG * HH, bhh + (size_t)l * 2 * GG,
          gxbuf, xout, hn_out, bs, s, (s == TT - 1) ? 1 : 0, l);
    }
    _Float16* t = xin; xin = xout; xout = t;
  }

  // 4) pack final output (xin holds last layer's output after swap)
  pack_kernel<<<TB, 256, 0, stream>>>(xin, bs, off, out);
}
